// TPRAttention_43903155700266
// MI455X (gfx1250) — matrix operand order repaired and checked by reading, not compiled, not run on hardware
//
#include <hip/hip_runtime.h>
#include <math.h>

#define BB 256
#define LL 512
#define DD 1024
#define K3 3072
#define CH 2              // context rows per TDM chunk
#define NCH (LL / CH)     // 256 chunks

typedef __attribute__((ext_vector_type(16))) __bf16 v16bf;
typedef __attribute__((ext_vector_type(8)))  float  v8f;
typedef __attribute__((ext_vector_type(4)))  unsigned int v4u;
typedef __attribute__((ext_vector_type(8)))  int v8i;
typedef __attribute__((ext_vector_type(4)))  int v4i;

union BF16x16 { v16bf v; __bf16 e[16]; };

__device__ __forceinline__ void cvt4(BF16x16& t, int i, float4 x) {
    t.e[i + 0] = (__bf16)x.x;
    t.e[i + 1] = (__bf16)x.y;
    t.e[i + 2] = (__bf16)x.z;
    t.e[i + 3] = (__bf16)x.w;
}

// A operand tile (16x32 bf16, M x K): lane&15 = row M.
// lanes 0-15 : elems 0-7 = K k..k+7,    elems 8-15 = K k+16..k+23
// lanes 16-31: elems 0-7 = K k+8..k+15, elems 8-15 = K k+24..k+31
__device__ __forceinline__ v16bf load_tile_A(const float* __restrict__ base,
                                             int ld, int row, int k, int lane) {
    const float* p = base + (size_t)row * ld + k + ((lane >> 4) << 3);
    BF16x16 t;
    cvt4(t, 0,  *(const float4*)(p));
    cvt4(t, 4,  *(const float4*)(p + 4));
    cvt4(t, 8,  *(const float4*)(p + 16));
    cvt4(t, 12, *(const float4*)(p + 20));
    return t.v;
}

// B operand tile (32x16 bf16, K x N). Column n = row n of W (computing X @ W^T).
// lanes 0-15 : column lane,    elems 0-7 = K k..k+7,    elems 8-15 = K k+16..k+23
// lanes 16-31: column lane-16, elems 0-7 = K k+8..k+15, elems 8-15 = K k+24..k+31
__device__ __forceinline__ v16bf load_tile_B(const float* __restrict__ base,
                                             int ld, int row, int k, int lane) {
    const float* p = base + (size_t)row * ld + k + ((lane >> 4) << 3);
    BF16x16 t;
    cvt4(t, 0,  *(const float4*)(p));
    cvt4(t, 4,  *(const float4*)(p + 4));
    cvt4(t, 8,  *(const float4*)(p + 16));
    cvt4(t, 12, *(const float4*)(p + 20));
    return t.v;
}

// C = A(MxK) @ W(NxK)^T. One 16x64 strip per wave: A operand converted once,
// fed to 4 WMMAs per K-step (K-step = 32).
__global__ void __launch_bounds__(128)
gemm_bf16_wmma(const float* __restrict__ A, const float* __restrict__ W,
               float* __restrict__ C, int ntn4, int K,
               int lda, int ldw, int ldc, int do_tanh) {
    int wave = blockIdx.x * 4 + (threadIdx.x >> 5);
    int lane = threadIdx.x & 31;
    int tm = wave / ntn4;
    int tn = wave % ntn4;            // strip of 64 columns
    int arow  = tm * 16 + (lane & 15);
    int wrow0 = tn * 64 + (lane & 15);

    v8f acc0 = {}, acc1 = {}, acc2 = {}, acc3 = {};
    for (int k = 0; k < K; k += 32) {
        v16bf a  = load_tile_A(A, lda, arow, k, lane);
        v16bf b0 = load_tile_B(W, ldw, wrow0,      k, lane);
        v16bf b1 = load_tile_B(W, ldw, wrow0 + 16, k, lane);
        v16bf b2 = load_tile_B(W, ldw, wrow0 + 32, k, lane);
        v16bf b3 = load_tile_B(W, ldw, wrow0 + 48, k, lane);
        acc0 = __builtin_amdgcn_wmma_f32_16x16x32_bf16(false, a, false, b0, (short)0, acc0, false, false);
        acc1 = __builtin_amdgcn_wmma_f32_16x16x32_bf16(false, a, false, b1, (short)0, acc1, false, false);
        acc2 = __builtin_amdgcn_wmma_f32_16x16x32_bf16(false, a, false, b2, (short)0, acc2, false, false);
        acc3 = __builtin_amdgcn_wmma_f32_16x16x32_bf16(false, a, false, b3, (short)0, acc3, false, false);
    }

    // D layout: VGPR r: lanes 0-15 -> M=r, N=lane; lanes 16-31 -> M=r+8, N=lane-16
    int mbase = tm * 16 + ((lane >> 4) << 3);
    v8f accs[4] = {acc0, acc1, acc2, acc3};
#pragma unroll
    for (int t = 0; t < 4; ++t) {
        int n = tn * 64 + t * 16 + (lane & 15);
#pragma unroll
        for (int r = 0; r < 8; ++r) {
            float v = accs[t][r];
            if (do_tanh) {
                float e = __expf(2.f * v);
                v = 1.f - 2.f / (e + 1.f);   // robust tanh: saturates to +/-1
            }
            C[(size_t)(mbase + r) * ldc + n] = v;
        }
    }
}

__device__ __forceinline__ float block_reduce_max(float v, float* red, int tid) {
#pragma unroll
    for (int off = 16; off; off >>= 1) v = fmaxf(v, __shfl_xor(v, off, 32));
    if ((tid & 31) == 0) red[tid >> 5] = v;
    __syncthreads();
    if (tid < 8) {
        float x = red[tid];
#pragma unroll
        for (int off = 4; off; off >>= 1) x = fmaxf(x, __shfl_xor(x, off, 8));
        if (tid == 0) red[0] = x;
    }
    __syncthreads();
    float r = red[0];
    __syncthreads();
    return r;
}

__device__ __forceinline__ float block_reduce_sum(float v, float* red, int tid) {
#pragma unroll
    for (int off = 16; off; off >>= 1) v += __shfl_xor(v, off, 32);
    if ((tid & 31) == 0) red[tid >> 5] = v;
    __syncthreads();
    if (tid < 8) {
        float x = red[tid];
#pragma unroll
        for (int off = 4; off; off >>= 1) x += __shfl_xor(x, off, 8);
        if (tid == 0) red[0] = x;
    }
    __syncthreads();
    float r = red[0];
    __syncthreads();
    return r;
}

// Issue one TDM 2D-tile load (CH rows x DD f32) from global into LDS.
// D# layout per CDNA5 ISA ch.8; amdgpu-toolchain 6-arg builtin form.
__device__ __forceinline__ void tdm_load_rows(const float* gptr,
                                              unsigned lds_off_bytes, int rows) {
    unsigned long long ga = (unsigned long long)(size_t)gptr;
    v4u g0;
    g0[0] = 1u;                                        // count=1, user mode
    g0[1] = lds_off_bytes;                             // lds_addr
    g0[2] = (unsigned)(ga & 0xFFFFFFFFu);              // global_addr[31:0]
    g0[3] = (unsigned)((ga >> 32) & 0x01FFFFFFu)       // global_addr[56:32]
          | (2u << 30);                                // type=2 ("image")
    v8i g1;
    g1[0] = 0x20000;                                   // data_size=2 (4 bytes)
    g1[1] = (int)((unsigned)DD << 16);                 // tensor_dim0[15:0]=1024
    g1[2] = (int)((unsigned)LL << 16);                 // tensor_dim1[15:0]=512
    g1[3] = (int)((unsigned)DD << 16);                 // tile_dim0=1024
    g1[4] = rows & 0xFFFF;                             // tile_dim1=rows, tile_dim2=0
    g1[5] = DD;                                        // tensor_dim0_stride=1024
    g1[6] = 0;
    g1[7] = 0;
    v4i z4 = {0, 0, 0, 0};
    v8i z8 = {0, 0, 0, 0, 0, 0, 0, 0};
    __builtin_amdgcn_tensor_load_to_lds(g0, g1, z4, z4, z8, 0);
}

// One workgroup (256 threads = 8 wave32) per batch row.
__global__ void __launch_bounds__(256)
attn_kernel(const float* __restrict__ input,
            const float* __restrict__ ctxF,
            const float* __restrict__ ctxR,
            const float* __restrict__ target,
            float* __restrict__ attnF_out,
            float* __restrict__ attnR_out,
            float* __restrict__ H) {
    __shared__ float s_t[DD];
    __shared__ float sF[LL];
    __shared__ float sR[LL];
    __shared__ float red[8];
    __shared__ float bufF[2][CH * DD];   // TDM double buffers
    __shared__ float bufR[2][CH * DD];

    const int b = blockIdx.x;
    const int tid = threadIdx.x;
    const int wave = tid >> 5, lane = tid & 31;

    const float* cF = ctxF + (size_t)b * LL * DD;
    const float* cR = ctxR + (size_t)b * LL * DD;

    // Stage target row in LDS
    const float4* tg4 = (const float4*)(target + (size_t)b * DD);
    for (int i = tid; i < DD / 4; i += 256) ((float4*)s_t)[i] = tg4[i];
    __syncthreads();

    // Pass 1: scores for F and R in one sweep of each row (one row per wave)
    for (int l = wave; l < LL; l += 8) {
        const float4* rF = (const float4*)(cF + (size_t)l * DD);
        const float4* rR = (const float4*)(cR + (size_t)l * DD);
        if (l + 8 < LL) {
            __builtin_prefetch(cF + (size_t)(l + 8) * DD + lane * 16, 0, 1);
            __builtin_prefetch(cR + (size_t)(l + 8) * DD + lane * 16, 0, 1);
        }
        float aF = 0.f, aR = 0.f;
        for (int j = lane; j < DD / 4; j += 32) {
            float4 t4 = ((const float4*)s_t)[j];
            float4 f4 = rF[j];
            float4 r4 = rR[j];
            aF += f4.x * t4.x + f4.y * t4.y + f4.z * t4.z + f4.w * t4.w;
            aR += r4.x * t4.x + r4.y * t4.y + r4.z * t4.z + r4.w * t4.w;
        }
#pragma unroll
        for (int off = 16; off; off >>= 1) {
            aF += __shfl_xor(aF, off, 32);
            aR += __shfl_xor(aR, off, 32);
        }
        if (lane == 0) { sF[l] = aF; sR[l] = aR; }
    }
    __syncthreads();

    // Softmax over L=512 (each thread owns elements tid and tid+256)
    float f0 = sF[tid], f1 = sF[tid + 256];
    float r0 = sR[tid], r1 = sR[tid + 256];

    float mF = block_reduce_max(fmaxf(f0, f1), red, tid);
    float e0 = __expf(f0 - mF), e1 = __expf(f1 - mF);
    float sumF = block_reduce_sum(e0 + e1, red, tid);
    float invF = __frcp_rn(sumF);
    e0 *= invF; e1 *= invF;

    float mR = block_reduce_max(fmaxf(r0, r1), red, tid);
    float g0 = __expf(r0 - mR), g1 = __expf(r1 - mR);
    float sumR = block_reduce_sum(g0 + g1, red, tid);
    float invR = __frcp_rn(sumR);
    g0 *= invR; g1 *= invR;

    sF[tid] = e0; sF[tid + 256] = e1;  // attnF reused as weights
    attnF_out[(size_t)b * LL + tid]       = e0;
    attnF_out[(size_t)b * LL + tid + 256] = e1;
    attnR_out[(size_t)b * LL + tid]       = g0;
    attnR_out[(size_t)b * LL + tid + 256] = g1;
    __syncthreads();

    // Pass 2: weighted sums (both use attnF, matching the reference).
    // Chunks of CH rows DMA'd into LDS via TDM, double-buffered, walked in
    // REVERSE row order so the tail of pass 1 is still hot in L2.
    const unsigned offF0 = (unsigned)(size_t)&bufF[0][0];
    const unsigned offF1 = (unsigned)(size_t)&bufF[1][0];
    const unsigned offR0 = (unsigned)(size_t)&bufR[0][0];
    const unsigned offR1 = (unsigned)(size_t)&bufR[1][0];

    if (tid < 32) {  // wave 0 drives the TDM pipeline
        tdm_load_rows(cF + (size_t)(NCH - 1) * CH * DD, offF0, CH);
        tdm_load_rows(cR + (size_t)(NCH - 1) * CH * DD, offR0, CH);
    }

    float4 aF = {0.f, 0.f, 0.f, 0.f};
    float4 aR = {0.f, 0.f, 0.f, 0.f};
    const int col = tid * 4;

    for (int i = 0; i < NCH; ++i) {
        const int c   = NCH - 1 - i;   // chunk (row block) being computed
        const int cur = i & 1;
        if (tid < 32) {
            if (i + 1 < NCH) {
                const int cn = c - 1;
                tdm_load_rows(cF + (size_t)cn * CH * DD, cur ? offF0 : offF1, CH);
                tdm_load_rows(cR + (size_t)cn * CH * DD, cur ? offR0 : offR1, CH);
                __builtin_amdgcn_s_wait_tensorcnt(2);  // chunk i's 2 ops done
            } else {
                __builtin_amdgcn_s_wait_tensorcnt(0);
            }
        }
        __syncthreads();               // chunk c visible in buf[cur] to all waves
#pragma unroll
        for (int r = 0; r < CH; ++r) {
            float w = sF[c * CH + r];
            float4 f4 = *(const float4*)&bufF[cur][r * DD + col];
            float4 r4 = *(const float4*)&bufR[cur][r * DD + col];
            aF.x += w * f4.x; aF.y += w * f4.y; aF.z += w * f4.z; aF.w += w * f4.w;
            aR.x += w * r4.x; aR.y += w * r4.y; aR.z += w * r4.z; aR.w += w * r4.w;
        }
        __syncthreads();               // done reading buf[cur] before it is refilled
    }

    // H row = [weighted_F | weighted_R | input]
    float4* Hrow = (float4*)(H + (size_t)b * K3);
    Hrow[tid]       = aF;
    Hrow[256 + tid] = aR;
    Hrow[512 + tid] = ((const float4*)(input + (size_t)b * DD))[tid];
}

extern "C" void kernel_launch(void* const* d_in, const int* in_sizes, int n_in,
                              void* d_out, int out_size, void* d_ws, size_t ws_size,
                              hipStream_t stream) {
    const float* input = (const float*)d_in[0];
    const float* ctxF  = (const float*)d_in[1];
    const float* ctxR  = (const float*)d_in[2];
    const float* W_in  = (const float*)d_in[3];
    const float* W_out = (const float*)d_in[4];

    float* out     = (float*)d_out;
    float* h_tilde = out;                            // [B, D]
    float* attnF   = out + (size_t)BB * DD;          // [B, L]
    float* attnR   = attnF + (size_t)BB * LL;        // [B, L]

    float* target = (float*)d_ws;                    // [B, D]
    float* H      = target + (size_t)BB * DD;        // [B, 3D]

    // GEMM1: target = input @ W_in^T   (M=256, N=1024, K=1024)
    {
        int waves = (BB / 16) * (DD / 64);           // 256 waves
        gemm_bf16_wmma<<<dim3(waves / 4), dim3(128), 0, stream>>>(
            input, W_in, target, DD / 64, DD, DD, DD, DD, 0);
    }

    // Attention: scores, softmax, TDM-pipelined weighted sums, build H
    attn_kernel<<<dim3(BB), dim3(256), 0, stream>>>(
        input, ctxF, ctxR, target, attnF, attnR, H);

    // GEMM2: h_tilde = tanh(H @ W_out^T)  (M=256, N=1024, K=3072)
    {
        int waves = (BB / 16) * (DD / 64);
        gemm_bf16_wmma<<<dim3(waves / 4), dim3(128), 0, stream>>>(
            H, W_out, h_tilde, DD / 64, K3, K3, K3, DD, 1);
    }
}
